// MultiEmbeder_69002944577623
// MI455X (gfx1250) — matrix operand order repaired and checked by reading, not compiled, not run on hardware
//
#include <hip/hip_runtime.h>

// ---------------------------------------------------------------------------
// MultiEmbeder loss on MI455X (gfx1250, wave32, WMMA bf16).
// Fused flash-softmax attention pooling: single HBM pass over each feature
// tensor (~235 MB total -> ~10us at 23.3 TB/s).
//  - chunk staged HBM -> LDS via GLOBAL_LOAD_ASYNC_TO_LDS_B128 (ASYNCcnt)
//  - score GEMM on v_wmma_f32_16x16x32_bf16 (A from LDS, B pre-packed in L2)
//  - tanh/exp via native TRANS32 ops (v_tanh_f32 / v_exp_f32), branch-free
// ---------------------------------------------------------------------------

#define NEGV  (-1e9f)
#define XSTF  260   // padded LDS row stride (f32) -> bank-rotating A-frag reads

typedef __attribute__((ext_vector_type(16))) __bf16 v16bf;
typedef __attribute__((ext_vector_type(8)))  float  v8f;

// async-to-LDS builtin pointee type (from toolchain diagnostic):
// arg0: v4i in global AS ("__device__"), arg1: LDS pointer
typedef __attribute__((__vector_size__(4 * sizeof(int)))) int vi4;
typedef __attribute__((address_space(1))) vi4* gvi4p;
typedef __attribute__((address_space(3))) vi4* lvi4p;

static __device__ __forceinline__ unsigned short f2bf(float x) {
  union { float f; unsigned u; } v; v.f = x;
  return (unsigned short)((v.u + 0x7FFFu + ((v.u >> 16) & 1u)) >> 16);  // RNE
}
static __device__ __forceinline__ unsigned pack2bf(float lo, float hi) {
  return (unsigned)f2bf(lo) | ((unsigned)f2bf(hi) << 16);
}

// Native CDNA5 transcendentals: branch-free, keeps EXEC uniform around WMMA.
static __device__ __forceinline__ float fast_tanh(float x) {
  float r;
  asm("v_tanh_f32 %0, %1\n\ts_delay_alu instid0(TRANS32_DEP_1)" : "=v"(r) : "v"(x));
  return r;
}
static __device__ __forceinline__ float fast_exp(float x) {
  return __builtin_amdgcn_exp2f(x * 1.44269504088896340736f);  // v_exp_f32
}

static __device__ __forceinline__ void wait_async0() {
#if __has_builtin(__builtin_amdgcn_s_wait_asynccnt)
  __builtin_amdgcn_s_wait_asynccnt(0);
#else
  asm volatile("s_wait_asynccnt 0x0" ::: "memory");
#endif
}

// ---------------------------------------------------------------------------
// Pre-pack a 256x256 f32 weight into per-lane WMMA B-fragment layout (bf16).
// Fragment f = nt*8 + kt covers B tile (K = kt*32..+31, N = nt*16..+15).
// 16-bit B 32x16 layout: lane l -> N = l&15; K = (l<16?0:8)+(e&7)+((e&8)?16:0).
// Stored lane-contiguous: frag*512 + lane*16 + e  (1 KB per fragment).
// ---------------------------------------------------------------------------
__global__ __launch_bounds__(256) void prep_wfrag(const float* __restrict__ W,
                                                  unsigned short* __restrict__ out) {
  int idx  = blockIdx.x * 256 + threadIdx.x;   // 0..65535
  int e    = idx & 15;
  int lane = (idx >> 4) & 31;
  int f    = idx >> 9;
  int kt = f & 7, nt = f >> 3;
  int k = kt * 32 + ((lane < 16) ? 0 : 8) + (e & 7) + ((e & 8) ? 16 : 0);
  int n = nt * 16 + (lane & 15);
  out[idx] = f2bf(W[k * 256 + n]);
}

// ---------------------------------------------------------------------------
// Fused attention-pool chunk kernel. grid = (nchunks, B), block = 256 (8 waves).
// Per chunk of 128 rows:   s_i = v . tanh(x_i W + b) + c  (masked -> NEG)
// writes partial record [m, l, acc[256]] with l = sum exp(s-m),
// acc = sum exp(s_i-m) * x_i  (flash-softmax partials).
// ---------------------------------------------------------------------------
__global__ __launch_bounds__(256) void pool_kernel(
    const float* __restrict__ feat, const unsigned char* __restrict__ mask,
    const unsigned short* __restrict__ Wfrag, const float* __restrict__ bvec,
    const float* __restrict__ vvec, const float* __restrict__ cscl,
    int S, int applyTanh, float* __restrict__ partials) {
  __shared__ float Xsf[128 * XSTF];
  __shared__ float bS[256], vS[256], sc[128], ev[128], red[256];
  __shared__ float cS;

  const int tid   = threadIdx.x;
  const int b     = blockIdx.y;
  const int chunk = blockIdx.x;
  const int r0    = chunk * 128;

  bS[tid] = bvec[tid];
  vS[tid] = vvec[tid];
  if (tid == 0) cS = cscl[0];

  // ---- stage 128x256 f32 chunk into LDS: async direct-to-LDS --------------
  const float* fbase = feat + ((size_t)b * S + r0) * 256;
#if __has_builtin(__builtin_amdgcn_global_load_async_to_lds_b128)
  for (int i = tid; i < 128 * 64; i += 256) {       // 32 x b128 per thread in flight
    int r = i >> 6, c4 = i & 63;
    __builtin_amdgcn_global_load_async_to_lds_b128(
        (gvi4p)(fbase + (size_t)r * 256 + c4 * 4),
        (lvi4p)&Xsf[r * XSTF + c4 * 4],
        0, 0);
  }
  wait_async0();
#else
  for (int i = tid; i < 128 * 64; i += 256) {
    int r = i >> 6, c4 = i & 63;
    float4 x = reinterpret_cast<const float4*>(fbase + (size_t)r * 256)[c4];
    *reinterpret_cast<float4*>(&Xsf[r * XSTF + c4 * 4]) = x;
  }
#endif
  __syncthreads();

  // ---- WMMA score GEMM: each wave owns a 16-row slab ----------------------
  const int lane = tid & 31, wave = tid >> 5;
  const int slab = wave * 16;
  const int mrow = lane & 15;
  const int ko   = (lane < 16) ? 0 : 8;   // 16-bit A 16x32 interleave

  union FragU { v16bf v; uint4 q[2]; unsigned u[8]; };
  FragU A[8];
  const int arow = slab + mrow;
#pragma unroll
  for (int kt = 0; kt < 8; ++kt) {
    const float* rp = &Xsf[arow * XSTF + kt * 32];
    float4 x0 = *reinterpret_cast<const float4*>(rp + ko);           // K = kt*32+ko..+3
    float4 x1 = *reinterpret_cast<const float4*>(rp + ko + 4);       // +4..+7
    float4 x2 = *reinterpret_cast<const float4*>(rp + 16 + ko);      // K = kt*32+16+ko..
    float4 x3 = *reinterpret_cast<const float4*>(rp + 16 + ko + 4);
    if (applyTanh) {
      x0.x = fast_tanh(x0.x); x0.y = fast_tanh(x0.y); x0.z = fast_tanh(x0.z); x0.w = fast_tanh(x0.w);
      x1.x = fast_tanh(x1.x); x1.y = fast_tanh(x1.y); x1.z = fast_tanh(x1.z); x1.w = fast_tanh(x1.w);
      x2.x = fast_tanh(x2.x); x2.y = fast_tanh(x2.y); x2.z = fast_tanh(x2.z); x2.w = fast_tanh(x2.w);
      x3.x = fast_tanh(x3.x); x3.y = fast_tanh(x3.y); x3.z = fast_tanh(x3.z); x3.w = fast_tanh(x3.w);
    }
    A[kt].u[0] = pack2bf(x0.x, x0.y);  A[kt].u[1] = pack2bf(x0.z, x0.w);
    A[kt].u[2] = pack2bf(x1.x, x1.y);  A[kt].u[3] = pack2bf(x1.z, x1.w);
    A[kt].u[4] = pack2bf(x2.x, x2.y);  A[kt].u[5] = pack2bf(x2.z, x2.w);
    A[kt].u[6] = pack2bf(x3.x, x3.y);  A[kt].u[7] = pack2bf(x3.z, x3.w);
  }

  float part[8];
#pragma unroll
  for (int j = 0; j < 8; ++j) part[j] = 0.f;

  const uint4* wq = reinterpret_cast<const uint4*>(Wfrag);
  for (int nt = 0; nt < 16; ++nt) {
    int   n  = nt * 16 + mrow;
    float bb = bS[n], vv = vS[n];
    v8f C = {0.f, 0.f, 0.f, 0.f, 0.f, 0.f, 0.f, 0.f};
#pragma unroll
    for (int kt = 0; kt < 8; ++kt) {
      FragU Bf;
      int qi = (nt * 8 + kt) * 64 + lane * 2;   // coalesced 2 x b128 per lane, L2-hot
      Bf.q[0] = wq[qi];
      Bf.q[1] = wq[qi + 1];
      C = __builtin_amdgcn_wmma_f32_16x16x32_bf16(false, A[kt].v, false, Bf.v,
                                                  (short)0, C, false, false);
    }
    // C: VGPR j, lane l -> (M = j + (l<16?0:8), N = n); fold tanh(.)+b, dot v
#pragma unroll
    for (int j = 0; j < 8; ++j) part[j] += fast_tanh(C[j] + bb) * vv;
  }

  // reduce over the 16 N-lanes of each half-wave
#pragma unroll
  for (int j = 0; j < 8; ++j) {
    part[j] += __shfl_xor(part[j], 1, 32);
    part[j] += __shfl_xor(part[j], 2, 32);
    part[j] += __shfl_xor(part[j], 4, 32);
    part[j] += __shfl_xor(part[j], 8, 32);
  }
  if (lane == 0 || lane == 16) {
    int rb = slab + ((lane == 16) ? 8 : 0);
#pragma unroll
    for (int j = 0; j < 8; ++j) {
      int  lr    = rb + j;
      bool valid = mask[(size_t)b * S + r0 + lr] != 0;
      sc[lr] = valid ? (part[j] + cS) : NEGV;
    }
  }
  __syncthreads();

  // ---- chunk softmax partials --------------------------------------------
  red[tid] = (tid < 128) ? sc[tid] : NEGV;
  __syncthreads();
  for (int s = 128; s >= 1; s >>= 1) {
    if (tid < s) red[tid] = fmaxf(red[tid], red[tid + s]);
    __syncthreads();
  }
  float m = red[0];
  __syncthreads();
  float e = 0.f;
  if (tid < 128) { e = fast_exp(sc[tid] - m); ev[tid] = e; }
  red[tid] = (tid < 128) ? e : 0.f;
  __syncthreads();
  for (int s = 128; s >= 1; s >>= 1) {
    if (tid < s) red[tid] += red[tid + s];
    __syncthreads();
  }
  float l = red[0];

  // weighted feature accumulation straight from the LDS-resident f32 chunk
  float acc = 0.f;
  if (applyTanh) {
    for (int i = 0; i < 128; ++i) acc += ev[i] * fast_tanh(Xsf[i * XSTF + tid]);
  } else {
    for (int i = 0; i < 128; ++i) acc += ev[i] * Xsf[i * XSTF + tid];
  }

  float* rec = partials + ((size_t)b * gridDim.x + chunk) * 258;
  rec[2 + tid] = acc;
  if (tid == 0) { rec[0] = m; rec[1] = l; }
}

// ---------------------------------------------------------------------------
// Merge chunk partials (log-sum-exp), normalize, apply 1 or 2 tanh.
// ---------------------------------------------------------------------------
__global__ __launch_bounds__(256) void merge_kernel(const float* __restrict__ partials,
                                                    int nchunks, float* __restrict__ outP,
                                                    int ntanh) {
  int b = blockIdx.x, h = threadIdx.x;
  const float* base = partials + (size_t)b * nchunks * 258;
  float m = -3.0e38f;
  for (int c = 0; c < nchunks; ++c) m = fmaxf(m, base[c * 258]);
  float L = 0.f, acc = 0.f;
  for (int c = 0; c < nchunks; ++c) {
    float s = fast_exp(base[c * 258] - m);
    L   += base[c * 258 + 1] * s;
    acc += base[c * 258 + 2 + h] * s;
  }
  float val = acc / fmaxf(L, 1e-30f);
  for (int t = 0; t < ntanh; ++t) val = fast_tanh(val);
  outP[b * 256 + h] = val;
}

// code = tanh(concat(tok, gr) @ Wf + bf)   [64,512] @ [512,256]
__global__ __launch_bounds__(256) void code_kernel(const float* __restrict__ tok,
                                                   const float* __restrict__ gr,
                                                   const float* __restrict__ Wf,
                                                   const float* __restrict__ bfv,
                                                   float* __restrict__ code) {
  __shared__ float cat[512];
  int b = blockIdx.x, h = threadIdx.x;
  cat[h]       = tok[b * 256 + h];
  cat[256 + h] = gr[b * 256 + h];
  __syncthreads();
  float acc = bfv[h];
  for (int k = 0; k < 512; ++k) acc += cat[k] * Wf[k * 256 + h];
  code[b * 256 + h] = fast_tanh(acc);
}

static __device__ __forceinline__ float blockSum256(float v, float* red) {
  int t = threadIdx.x;
  red[t] = v; __syncthreads();
  for (int s = 128; s >= 1; s >>= 1) {
    if (t < s) red[t] += red[t + s];
    __syncthreads();
  }
  float r = red[0]; __syncthreads();
  return r;
}

__global__ __launch_bounds__(256) void loss_kernel(const float* __restrict__ code,
                                                   const float* __restrict__ da,
                                                   const float* __restrict__ dn,
                                                   float* __restrict__ hinge) {
  __shared__ float red[256];
  int b = blockIdx.x, h = threadIdx.x;
  float cv = code[b * 256 + h], av = da[b * 256 + h], nv = dn[b * 256 + h];
  float cc = blockSum256(cv * cv, red);
  float aa = blockSum256(av * av, red);
  float nn = blockSum256(nv * nv, red);
  float ca = blockSum256(cv * av, red);
  float cn = blockSum256(cv * nv, red);
  if (h == 0) {
    float ncd = sqrtf(cc), nad = sqrtf(aa), nnd = sqrtf(nn);
    float cosA = ca / fmaxf(ncd * nad, 1e-8f);
    float cosN = cn / fmaxf(ncd * nnd, 1e-8f);
    hinge[b] = fmaxf(0.6f - cosA + cosN, 0.f);
  }
}

__global__ void mean_kernel(const float* __restrict__ hinge, float* __restrict__ out) {
  if (threadIdx.x == 0) {
    float s = 0.f;
    for (int i = 0; i < 64; ++i) s += hinge[i];
    out[0] = s / 64.f;
  }
}

// ---------------------------------------------------------------------------
extern "C" void kernel_launch(void* const* d_in, const int* in_sizes, int n_in,
                              void* d_out, int out_size, void* d_ws, size_t ws_size,
                              hipStream_t stream) {
  (void)n_in; (void)out_size; (void)ws_size;
  const int B = 64, H = 256;

  const float*         tokF = (const float*)d_in[0];
  const unsigned char* tokM = (const unsigned char*)d_in[1];
  const float*         grF  = (const float*)d_in[2];
  const unsigned char* grM  = (const unsigned char*)d_in[3];
  const float*         daF  = (const float*)d_in[4];
  const unsigned char* daM  = (const unsigned char*)d_in[5];
  const float*         dnF  = (const float*)d_in[6];
  const unsigned char* dnM  = (const unsigned char*)d_in[7];
  const float *Wt = (const float*)d_in[8],  *bt = (const float*)d_in[9];
  const float *vt = (const float*)d_in[10], *ct = (const float*)d_in[11];
  const float *Wg = (const float*)d_in[12], *bg = (const float*)d_in[13];
  const float *vg = (const float*)d_in[14], *cg = (const float*)d_in[15];
  const float *Wd = (const float*)d_in[16], *bd = (const float*)d_in[17];
  const float *vd = (const float*)d_in[18], *cd = (const float*)d_in[19];
  const float *Wf = (const float*)d_in[20], *bfv = (const float*)d_in[21];

  int St = in_sizes[0] / (B * H);
  int Sg = in_sizes[2] / (B * H);
  int Sd = in_sizes[4] / (B * H);
  int chT = St / 128, chG = Sg / 128, chD = Sd / 128;

  char*  ws  = (char*)d_ws;
  size_t off = 0;
  auto take = [&](size_t bytes) {
    size_t o = off;
    off += (bytes + 255) & ~(size_t)255;
    return o;
  };
  unsigned short* WtF  = (unsigned short*)(ws + take(65536 * 2));
  unsigned short* WgF  = (unsigned short*)(ws + take(65536 * 2));
  unsigned short* WdF  = (unsigned short*)(ws + take(65536 * 2));
  float* pT    = (float*)(ws + take((size_t)B * chT * 258 * 4));
  float* pG    = (float*)(ws + take((size_t)B * chG * 258 * 4));
  float* pA    = (float*)(ws + take((size_t)B * chD * 258 * 4));
  float* pN    = (float*)(ws + take((size_t)B * chD * 258 * 4));
  float* tokP  = (float*)(ws + take((size_t)B * H * 4));
  float* grP   = (float*)(ws + take((size_t)B * H * 4));
  float* daP   = (float*)(ws + take((size_t)B * H * 4));
  float* dnP   = (float*)(ws + take((size_t)B * H * 4));
  float* codeP = (float*)(ws + take((size_t)B * H * 4));
  float* hingP = (float*)(ws + take((size_t)B * 4));

  prep_wfrag<<<256, 256, 0, stream>>>(Wt, WtF);
  prep_wfrag<<<256, 256, 0, stream>>>(Wg, WgF);
  prep_wfrag<<<256, 256, 0, stream>>>(Wd, WdF);

  pool_kernel<<<dim3(chT, B), 256, 0, stream>>>(tokF, tokM, WtF, bt, vt, ct, St, 0, pT);
  pool_kernel<<<dim3(chG, B), 256, 0, stream>>>(grF,  grM,  WgF, bg, vg, cg, Sg, 1, pG);
  pool_kernel<<<dim3(chD, B), 256, 0, stream>>>(daF,  daM,  WdF, bd, vd, cd, Sd, 0, pA);
  pool_kernel<<<dim3(chD, B), 256, 0, stream>>>(dnF,  dnM,  WdF, bd, vd, cd, Sd, 0, pN);

  merge_kernel<<<B, 256, 0, stream>>>(pT, chT, tokP, 1);
  merge_kernel<<<B, 256, 0, stream>>>(pG, chG, grP, 1);
  merge_kernel<<<B, 256, 0, stream>>>(pA, chD, daP, 2);
  merge_kernel<<<B, 256, 0, stream>>>(pN, chD, dnP, 2);

  code_kernel<<<B, 256, 0, stream>>>(tokP, grP, Wf, bfv, codeP);
  loss_kernel<<<B, 256, 0, stream>>>(codeP, daP, dnP, hingP);
  mean_kernel<<<1, 64, 0, stream>>>(hingP, (float*)d_out);
}
